// CausalSelfAttention_60146722013760
// MI455X (gfx1250) — matrix operand order repaired and checked by reading, compiled, not run on hardware
//
#include <hip/hip_runtime.h>
#include <hip/hip_bf16.h>

#define HD __device__ __forceinline__

typedef __attribute__((ext_vector_type(16))) __bf16 v16bf;
typedef __attribute__((ext_vector_type(8)))  float  v8f;

union BF16x16 {
  v16bf v;
  unsigned short u[16];
  unsigned int   d[8];
};

// B=2, T=2048, C=1024, H=16, D=64
#define BB 2
#define TT 2048
#define CC 1024
#define HH 16
#define DD 64
#define C3 3072
#define MM 4096 /* B*T */

HD unsigned short f2bf(float f) {
  unsigned int u = __float_as_uint(f);
  u += 0x7FFFu + ((u >> 16) & 1u);     // round-to-nearest-even
  return (unsigned short)(u >> 16);
}

HD v8f wmma_bf16(v16bf a, v16bf b, v8f c) {
  // D = A(16x32 bf16) x B(32x16 bf16) + C(16x16 f32)
  return __builtin_amdgcn_wmma_f32_16x16x32_bf16(
      /*neg_a=*/false, a, /*neg_b=*/false, b,
      /*c_mod=*/(short)0, c, /*reuse_a=*/false, /*reuse_b=*/false);
}

// ---------------------------------------------------------------------------
// Prep: dst[n*K + k] = bf16(src[k*N + n])   (weight transpose + convert)
// ---------------------------------------------------------------------------
__global__ void cvt_transpose_kernel(const float* __restrict__ src,
                                     unsigned short* __restrict__ dst,
                                     int K, int N) {
  int idx = blockIdx.x * 256 + threadIdx.x;
  if (idx >= K * N) return;
  int n = idx / K;
  int k = idx - n * K;
  dst[idx] = f2bf(src[k * N + n]);
}

// ---------------------------------------------------------------------------
// Kernel 1: QKV projection.  One wave computes a 16x64 tile of x @ w_attn.
// B fragments for all 4 N-tiles are batch-loaded before the 4 WMMAs so loads
// overlap matrix ops; K-loop unrolled 2x for software pipelining.
// Epilogue adds bias, pre-scales Q by 1/sqrt(D), and scatters:
//   Q,K -> bf16 [B,H,T,D],  V -> bf16 [B,H,D,T] (transposed for PV matmul)
// ---------------------------------------------------------------------------
__global__ __launch_bounds__(32) void qkv_kernel(
    const float* __restrict__ x, const unsigned short* __restrict__ wT,
    const float* __restrict__ bias, unsigned short* __restrict__ Q,
    unsigned short* __restrict__ Kd, unsigned short* __restrict__ Vt) {
  const int lane  = threadIdx.x & 31;
  const int lo    = lane & 15;
  const int hal8  = (lane >> 4) << 3;
  const int hal16 = (lane >> 4) << 4;
  const int m0 = blockIdx.x << 4;   // row in [0, 4096)
  const int n0 = blockIdx.y << 6;   // col in [0, 3072)

  v8f acc[4];
  for (int i = 0; i < 4; ++i)
    for (int j = 0; j < 8; ++j) acc[i][j] = 0.f;

  const float* xr = x + (size_t)(m0 + lo) * CC;
  const unsigned short* wr = wT + (size_t)(n0 + lo) * CC + hal16;

#pragma unroll 2
  for (int k0 = 0; k0 < CC; k0 += 32) {
    // batch-load all B fragments (8 x global_load_b128) first
    BF16x16 b[4];
#pragma unroll
    for (int nt = 0; nt < 4; ++nt)
      b[nt].v = *(const v16bf*)(wr + (size_t)nt * 16 * CC + k0);

    BF16x16 a;                       // A-fragment: row m0+lo, contiguous k = 16h+s
#pragma unroll
    for (int v = 0; v < 8; ++v) {
      int kk = hal16 + (v << 1);
      float2 f = *(const float2*)(xr + k0 + kk);
      a.u[2 * v]     = f2bf(f.x);
      a.u[2 * v + 1] = f2bf(f.y);
    }
#pragma unroll
    for (int nt = 0; nt < 4; ++nt)
      acc[nt] = wmma_bf16(a.v, b[nt].v, acc[nt]);
  }
#pragma unroll
  for (int nt = 0; nt < 4; ++nt) {
#pragma unroll
    for (int i = 0; i < 8; ++i) {
      int row = m0 + i + hal8;
      int col = n0 + nt * 16 + lo;
      float val = acc[nt][i] + bias[col];
      int bidx = row >> 11;          // row / T
      int t    = row & (TT - 1);
      int region = col >> 10;        // 0:Q 1:K 2:V
      int nl = col & (CC - 1);
      int h = nl >> 6, d = nl & 63;
      int bh = (bidx << 4) + h;
      if (region == 0)
        Q[((size_t)bh * TT + t) * DD + d] = f2bf(val * 0.125f);
      else if (region == 1)
        Kd[((size_t)bh * TT + t) * DD + d] = f2bf(val);
      else
        Vt[((size_t)bh * DD + d) * TT + t] = f2bf(val);
    }
  }
}

// ---------------------------------------------------------------------------
// Kernel 2: flash attention, one wave per (b*H+h, 16-query tile).
// Key blocks of 32: 4 WMMAs for S=Q*K^T, fp32 online softmax, P transposed
// C-layout -> A-layout through per-wave LDS, 4 WMMAs for O += P*V.
// K and V fragments batch-loaded before their WMMA groups.
// ---------------------------------------------------------------------------
__global__ __launch_bounds__(32) void attn_kernel(
    const unsigned short* __restrict__ Q, const unsigned short* __restrict__ Kd,
    const unsigned short* __restrict__ Vt, unsigned short* __restrict__ attnb) {
  __shared__ __align__(16) unsigned short plds[16 * 34];
  const int lane  = threadIdx.x & 31;
  const int lo    = lane & 15;
  const int hal8  = (lane >> 4) << 3;
  const int hal16 = (lane >> 4) << 4;
  const int q0 = blockIdx.x << 4;
  const int bh = blockIdx.y;

  const unsigned short* Qb = Q  + (size_t)bh * TT * DD;
  const unsigned short* Kb = Kd + (size_t)bh * TT * DD + (size_t)lo * DD + hal16;
  const unsigned short* Vb = Vt + (size_t)bh * DD * TT + (size_t)lo * TT + hal16;

  // Q A-fragments (two 16x32 K-chunks over D=64), loaded once per tile
  BF16x16 aq[2];
  {
    const unsigned short* qr = Qb + (size_t)(q0 + lo) * DD;
#pragma unroll
    for (int c = 0; c < 2; ++c)
#pragma unroll
      for (int v = 0; v < 8; ++v) {
        int dd = c * 32 + hal16 + (v << 1);
        aq[c].d[v] = *(const unsigned int*)(qr + dd);
      }
  }

  v8f o[4];
  for (int i = 0; i < 4; ++i)
    for (int j = 0; j < 8; ++j) o[i][j] = 0.f;
  float mrow[8], lrow[8];
  for (int i = 0; i < 8; ++i) { mrow[i] = -1e30f; lrow[i] = 0.f; }

  const float NEG = -1e30f;
  for (int k0 = 0; k0 <= q0 + 15; k0 += 32) {
    // batch-load the 4 K-fragments (2 key sub-tiles x 2 D-chunks)
    BF16x16 kb[4];
#pragma unroll
    for (int c = 0; c < 2; ++c) {
      kb[c * 2 + 0].v = *(const v16bf*)(Kb + (size_t)k0 * DD + c * 32);
      kb[c * 2 + 1].v = *(const v16bf*)(Kb + (size_t)(k0 + 16) * DD + c * 32);
    }
    v8f s0, s1;
    for (int j = 0; j < 8; ++j) { s0[j] = 0.f; s1[j] = 0.f; }
#pragma unroll
    for (int c = 0; c < 2; ++c) {
      s0 = wmma_bf16(aq[c].v, kb[c * 2 + 0].v, s0);
      s1 = wmma_bf16(aq[c].v, kb[c * 2 + 1].v, s1);
    }
    // batch-load the 4 V^T fragments while softmax runs on VALU
    BF16x16 vb[4];
#pragma unroll
    for (int nt = 0; nt < 4; ++nt)
      vb[nt].v = *(const v16bf*)(Vb + (size_t)nt * 16 * TT + k0);

    const int col0 = k0 + lo, col1 = k0 + 16 + lo;
    float p0[8], p1[8], alpha[8];
#pragma unroll
    for (int i = 0; i < 8; ++i) {
      const int rowg = q0 + i + hal8;
      float a0 = (col0 > rowg) ? NEG : s0[i];   // causal mask
      float a1 = (col1 > rowg) ? NEG : s1[i];
      float tm = fmaxf(a0, a1);
#pragma unroll
      for (int msk = 1; msk < 16; msk <<= 1)
        tm = fmaxf(tm, __shfl_xor(tm, msk, 16));
      float nm = fmaxf(mrow[i], tm);
      alpha[i] = __expf(mrow[i] - nm);
      p0[i] = __expf(a0 - nm);
      p1[i] = __expf(a1 - nm);
      float rs = p0[i] + p1[i];
#pragma unroll
      for (int msk = 1; msk < 16; msk <<= 1)
        rs += __shfl_xor(rs, msk, 16);
      lrow[i] = lrow[i] * alpha[i] + rs;
      mrow[i] = nm;
    }
#pragma unroll
    for (int nt = 0; nt < 4; ++nt)
#pragma unroll
      for (int i = 0; i < 8; ++i) o[nt][i] *= alpha[i];

    // C-layout -> A-layout transpose of P through LDS (per-wave slab; DS
    // ops from one wave are processed in order, no barrier needed)
#pragma unroll
    for (int i = 0; i < 8; ++i) {
      int r = i + hal8;
      plds[r * 34 + lo]      = f2bf(p0[i]);
      plds[r * 34 + 16 + lo] = f2bf(p1[i]);
    }
    BF16x16 pa;
#pragma unroll
    for (int v = 0; v < 8; ++v) {
      int kk = hal16 + (v << 1);
      pa.d[v] = *(const unsigned int*)&plds[lo * 34 + kk];
    }
#pragma unroll
    for (int nt = 0; nt < 4; ++nt)
      o[nt] = wmma_bf16(pa.v, vb[nt].v, o[nt]);
  }

  const int b = bh >> 4, h = bh & 15;
#pragma unroll
  for (int i = 0; i < 8; ++i) {
    float inv = 1.f / lrow[i];
    int t = q0 + i + hal8;
    unsigned short* orow = attnb + ((size_t)(b * TT + t)) * CC + h * DD;
#pragma unroll
    for (int nt = 0; nt < 4; ++nt)
      orow[nt * 16 + lo] = f2bf(o[nt][i] * inv);
  }
}

// ---------------------------------------------------------------------------
// Kernel 3: output projection, fp32 out = attn(bf16) @ w_proj + b_proj
// ---------------------------------------------------------------------------
__global__ __launch_bounds__(32) void proj_kernel(
    const unsigned short* __restrict__ attnb, const unsigned short* __restrict__ wT,
    const float* __restrict__ bias, float* __restrict__ out) {
  const int lane  = threadIdx.x & 31;
  const int lo    = lane & 15;
  const int hal8  = (lane >> 4) << 3;
  const int hal16 = (lane >> 4) << 4;
  const int m0 = blockIdx.x << 4;
  const int n0 = blockIdx.y << 6;

  v8f acc[4];
  for (int i = 0; i < 4; ++i)
    for (int j = 0; j < 8; ++j) acc[i][j] = 0.f;

  const unsigned short* ar = attnb + (size_t)(m0 + lo) * CC;
  const unsigned short* wr = wT + (size_t)(n0 + lo) * CC + hal16;

#pragma unroll 2
  for (int k0 = 0; k0 < CC; k0 += 32) {
    BF16x16 b[4];
#pragma unroll
    for (int nt = 0; nt < 4; ++nt)
      b[nt].v = *(const v16bf*)(wr + (size_t)nt * 16 * CC + k0);

    BF16x16 a;
#pragma unroll
    for (int v = 0; v < 8; ++v) {
      int kk = hal16 + (v << 1);
      a.d[v] = *(const unsigned int*)(ar + k0 + kk);
    }
#pragma unroll
    for (int nt = 0; nt < 4; ++nt)
      acc[nt] = wmma_bf16(a.v, b[nt].v, acc[nt]);
  }
#pragma unroll
  for (int nt = 0; nt < 4; ++nt)
#pragma unroll
    for (int i = 0; i < 8; ++i) {
      int row = m0 + i + hal8;
      int col = n0 + nt * 16 + lo;
      out[(size_t)row * CC + col] = acc[nt][i] + bias[col];
    }
}

// ---------------------------------------------------------------------------
extern "C" void kernel_launch(void* const* d_in, const int* in_sizes, int n_in,
                              void* d_out, int out_size, void* d_ws, size_t ws_size,
                              hipStream_t stream) {
  (void)in_sizes; (void)n_in; (void)out_size; (void)ws_size;
  const float* x      = (const float*)d_in[0];
  const float* w_attn = (const float*)d_in[1];
  const float* b_attn = (const float*)d_in[2];
  const float* w_proj = (const float*)d_in[3];
  const float* b_proj = (const float*)d_in[4];
  float* out = (float*)d_out;

  // workspace layout (bf16 elements): Q|K|Vt|attn|w_attn^T|w_proj^T = 40 MB
  unsigned short* ws    = (unsigned short*)d_ws;
  unsigned short* Q     = ws;
  unsigned short* Kd    = Q     + (size_t)BB * HH * TT * DD;
  unsigned short* Vt    = Kd    + (size_t)BB * HH * TT * DD;
  unsigned short* attnb = Vt    + (size_t)BB * HH * TT * DD;
  unsigned short* wAT   = attnb + (size_t)MM * CC;
  unsigned short* wPT   = wAT   + (size_t)C3 * CC;

  cvt_transpose_kernel<<<(C3 * CC + 255) / 256, 256, 0, stream>>>(w_attn, wAT, CC, C3);
  cvt_transpose_kernel<<<(CC * CC + 255) / 256, 256, 0, stream>>>(w_proj, wPT, CC, CC);
  qkv_kernel<<<dim3(MM / 16, C3 / 64), 32, 0, stream>>>(x, wAT, b_attn, Q, Kd, Vt);
  attn_kernel<<<dim3(TT / 16, BB * HH), 32, 0, stream>>>(Q, Kd, Vt, attnb);
  proj_kernel<<<dim3(MM / 16, CC / 64), 32, 0, stream>>>(attnb, wPT, b_proj, out);
}
